// RelationalTransformer_36034775614217
// MI455X (gfx1250) — matrix operand order repaired and checked by reading, not compiled, not run on hardware
//
#include <hip/hip_runtime.h>

// ---------------- CDNA5 (gfx1250) wave32 / WMMA helpers ----------------
typedef __attribute__((ext_vector_type(16))) __bf16 bf16x16;
typedef __attribute__((ext_vector_type(8)))  float  f32x8;
typedef __bf16 bf16_t;

union FragU { bf16x16 v; uint4 q[2]; };     // 32B fragment (16 x bf16)

__device__ inline f32x8 zero8() {
  f32x8 z;
  #pragma unroll
  for (int k = 0; k < 8; k++) z[k] = 0.f;
  return z;
}

// native hardware conversions (gfx1250 has bf16 cvt in the VALU)
__device__ inline bf16_t f2b(float f) { return (bf16_t)f; }
__device__ inline float  b2f(bf16_t b) { return (float)b; }

__device__ inline float gelu_f(float x) {
  return 0.5f * x * (1.0f + erff(x * 0.70710678118654752440f));
}

// dims
#define Bn   2
#define Nn   256
#define DNc  128
#define DEc  64
#define DHc  128
#define Hh   8
#define DNHc 512
#define DEHc 256
#define Lc   2
#define SCALE_ATTN 0.08838834764831845f  // 1/sqrt(128)

// ======================================================================
// f32 -> bf16 convert (row-major, for A-side activations)
// ======================================================================
__global__ void cvt_f32_bf16(const float* __restrict__ src, bf16_t* __restrict__ dst, int n) {
  int i = blockIdx.x * blockDim.x + threadIdx.x;
  if (i < n) dst[i] = f2b(src[i]);
}

// ======================================================================
// Repack weight (K x N, f32) into WMMA B-fragment order (bf16):
//   dst[ ((kt*N + n)*2 + half)*16 + t ] = src[ (kt*32 + (t&8)*2 + half*8 + (t&7))*N + n ]
// so lane l of a wave reads its 16 values as one contiguous 32B run.
// blockIdx.y = layer (src/dst strided by K*N).
// ======================================================================
__global__ void repack_b(const float* __restrict__ src, bf16_t* __restrict__ dst, int K, int N) {
  int o = blockIdx.x * blockDim.x + threadIdx.x;
  int total = K * N;
  if (o >= total) return;
  size_t base = (size_t)blockIdx.y * total;
  int t = o & 15, half = (o >> 4) & 1;
  int n = (o >> 5) % N;
  int kt = o / (N * 32);
  int k = kt * 32 + (t & 8) * 2 + half * 8 + (t & 7);
  dst[base + o] = f2b(src[base + (size_t)k * N + n]);
}

// ======================================================================
// Generic bf16 WMMA GEMM: C(MxN,f32) = A(MxK,bf16) @ Bp(frag-packed bf16)
// block = 128 threads = 4 waves (2x2), block tile 64x64, K-step 32.
// B fragments come straight from global (L2-resident repacked weights).
// EPI bit0 = add bias, bit1 = gelu.  WBF: also write bf16 shadow of C.
// ======================================================================
template <int EPI, bool WBF>
__global__ __launch_bounds__(128) void gemm_bf16(
    const bf16_t* __restrict__ A, const bf16_t* __restrict__ Bp,
    const float* __restrict__ bias, float* __restrict__ C, bf16_t* __restrict__ Cbf,
    int M, int N, int K) {
  __shared__ __align__(16) bf16_t As[64 * 32];
  int m0 = blockIdx.x * 64, n0 = blockIdx.y * 64;
  int tid = threadIdx.x, wave = tid >> 5, lane = tid & 31;
  int wr = wave >> 1, wc = wave & 1;
  int l15 = lane & 15, kh = (lane >> 4) * 8, rhalf = (lane >> 4) * 8, kb16 = (lane >> 4) * 16;

  f32x8 acc[2][2];
  #pragma unroll
  for (int i = 0; i < 2; i++)
    #pragma unroll
    for (int j = 0; j < 2; j++) acc[i][j] = zero8();

  for (int k0 = 0; k0 < K; k0 += 32) {
    __syncthreads();
    #pragma unroll
    for (int c = 0; c < 2; c++) {                 // A: 64x32 = 2048 elems
      int idx = tid * 16 + c * 8;
      int r = idx >> 5, col = idx & 31;
      *(uint4*)&As[idx] = *(const uint4*)(A + (size_t)(m0 + r) * K + k0 + col);
    }
    __syncthreads();

    int kt = k0 >> 5;
    FragU af[2], bfr[2];
    #pragma unroll
    for (int fi = 0; fi < 2; fi++) {
      int r = wr * 32 + fi * 16 + l15;
      af[fi].q[0] = *(const uint4*)&As[r * 32 + kh];
      af[fi].q[1] = *(const uint4*)&As[r * 32 + 16 + kh];
      int n = n0 + wc * 32 + fi * 16 + l15;
      const bf16_t* bp = Bp + ((size_t)kt * N + n) * 32 + kb16;
      bfr[fi].q[0] = *(const uint4*)bp;
      bfr[fi].q[1] = *(const uint4*)(bp + 8);
    }
    #pragma unroll
    for (int i = 0; i < 2; i++)
      #pragma unroll
      for (int j = 0; j < 2; j++)
        acc[i][j] = __builtin_amdgcn_wmma_f32_16x16x32_bf16(
            false, af[i].v, false, bfr[j].v, (short)0, acc[i][j], false, false);
  }

  #pragma unroll
  for (int i = 0; i < 2; i++)
    #pragma unroll
    for (int j = 0; j < 2; j++) {
      int gn = n0 + wc * 32 + j * 16 + l15;
      float bv = (EPI & 1) ? bias[gn] : 0.f;
      #pragma unroll
      for (int v = 0; v < 8; v++) {
        int gm = m0 + wr * 32 + i * 16 + rhalf + v;
        float val = acc[i][j][v] + bv;
        if (EPI & 2) val = gelu_f(val);
        C[(size_t)gm * N + gn] = val;
        if (WBF) Cbf[(size_t)gm * N + gn] = f2b(val);
      }
    }
}

// ======================================================================
// Fused attention: per-(b,i) workgroup; 8 waves = 8 heads.
// Each j-tile (32 rows): project e[b,i,jt,:] @ Wqkv_e (WMMA) -> per-head
// eq/ek/ev/em in LDS, then online (flash) softmax over j with wave32
// butterfly reductions.  qkv_e is never materialized in HBM.
// ======================================================================
__global__ __launch_bounds__(256) void attn_kernel(
    const bf16_t* __restrict__ ebf,      // (B,N,N,64) bf16 row-major
    const bf16_t* __restrict__ Wqkvep,   // (64,512) frag-packed (layer slice)
    const float*  __restrict__ qkvn,     // (B*N,384) f32
    float* __restrict__ outf, bf16_t* __restrict__ outbf) {  // (B*N,128)
  __shared__ __align__(16) bf16_t Ps[Hh * 32 * 64];
  int bi = blockIdx.x; int b = bi >> 8; int i = bi & 255;
  int tid = threadIdx.x, h = tid >> 5, lane = tid & 31;
  int l15 = lane & 15, kh = (lane >> 4) * 8, rhalf = (lane >> 4) * 8, kb16 = (lane >> 4) * 16;

  float qv[16];
  const float* qrow = qkvn + (size_t)(b * Nn + i) * 384 + h * 48;
  #pragma unroll
  for (int d = 0; d < 16; d++) qv[d] = qrow[d];

  float m_run = -3.0e38f, lsum = 0.f, acc[16];
  #pragma unroll
  for (int d = 0; d < 16; d++) acc[d] = 0.f;
  bf16_t* myPs = Ps + h * 32 * 64;

  for (int jt = 0; jt < 8; jt++) {
    int j0 = jt * 32;
    if (jt < 7)   // prefetch next tile's e rows (gfx1250 global_prefetch_b8)
      __builtin_prefetch(ebf + ((size_t)((b * Nn + i) * Nn + j0 + 32 + lane)) * 64, 0, 3);

    f32x8 pacc[2][4];
    #pragma unroll
    for (int a = 0; a < 2; a++)
      #pragma unroll
      for (int c = 0; c < 4; c++) pacc[a][c] = zero8();

    #pragma unroll
    for (int ks = 0; ks < 64; ks += 32) {
      FragU af[2];
      #pragma unroll
      for (int fi = 0; fi < 2; fi++) {
        int j = j0 + fi * 16 + l15;
        const bf16_t* arow = ebf + ((size_t)((b * Nn + i) * Nn + j)) * 64 + ks;
        af[fi].q[0] = *(const uint4*)(arow + kh);
        af[fi].q[1] = *(const uint4*)(arow + 16 + kh);
      }
      #pragma unroll
      for (int fj = 0; fj < 4; fj++) {
        FragU bfr;
        int col = h * 64 + fj * 16 + l15;
        const bf16_t* bp = Wqkvep + ((size_t)(ks >> 5) * 512 + col) * 32 + kb16;
        bfr.q[0] = *(const uint4*)bp;
        bfr.q[1] = *(const uint4*)(bp + 8);
        #pragma unroll
        for (int fi = 0; fi < 2; fi++)
          pacc[fi][fj] = __builtin_amdgcn_wmma_f32_16x16x32_bf16(
              false, af[fi].v, false, bfr.v, (short)0, pacc[fi][fj], false, false);
      }
    }
    // spill this head's 32x64 projection tile to LDS (wave-private region)
    #pragma unroll
    for (int fi = 0; fi < 2; fi++)
      #pragma unroll
      for (int fj = 0; fj < 4; fj++)
        #pragma unroll
        for (int v = 0; v < 8; v++)
          myPs[(fi * 16 + rhalf + v) * 64 + fj * 16 + l15] = f2b(pacc[fi][fj][v]);

    // flash update: lane <-> j = j0+lane; read whole 64-elem row as 8x b128
    int j = j0 + lane;
    const float* krow = qkvn + (size_t)(b * Nn + j) * 384 + h * 48;
    union { uint4 q[8]; bf16_t e[64]; } pr;
    const uint4* pq = (const uint4*)(myPs + lane * 64);
    #pragma unroll
    for (int q8 = 0; q8 < 8; q8++) pr.q[q8] = pq[q8];

    float evv[16], emm[16], dot = 0.f;
    #pragma unroll
    for (int d = 0; d < 16; d++) {
      float eq = b2f(pr.e[d]), ek = b2f(pr.e[16 + d]);
      evv[d] = b2f(pr.e[32 + d]); emm[d] = b2f(pr.e[48 + d]);
      dot += (qv[d] + eq) * (krow[16 + d] + ek);
    }
    dot *= SCALE_ATTN;
    float tmax = dot;
    #pragma unroll
    for (int s = 16; s >= 1; s >>= 1) tmax = fmaxf(tmax, __shfl_xor(tmax, s, 32));
    float m_new = fmaxf(m_run, tmax);
    float corr = expf(m_run - m_new);
    float p = expf(dot - m_new);
    float psum = p;
    #pragma unroll
    for (int s = 16; s >= 1; s >>= 1) psum += __shfl_xor(psum, s, 32);
    lsum = lsum * corr + psum;
    #pragma unroll
    for (int d = 0; d < 16; d++) {
      float val = p * (krow[32 + d] * emm[d] + evv[d]);
      #pragma unroll
      for (int s = 16; s >= 1; s >>= 1) val += __shfl_xor(val, s, 32);
      acc[d] = acc[d] * corr + val;
    }
    m_run = m_new;
  }
  if (lane == 0) {
    float inv = 1.f / lsum;
    size_t o = (size_t)(b * Nn + i) * 128 + h * 16;
    #pragma unroll
    for (int d = 0; d < 16; d++) { float v = acc[d] * inv; outf[o + d] = v; outbf[o + d] = f2b(v); }
  }
}

// ======================================================================
// node residual + LayerNorm (row = 128), one wave per row
// ======================================================================
__global__ __launch_bounds__(256) void addln_node(
    const float* __restrict__ xin, const float* __restrict__ t,
    const float* __restrict__ g, const float* __restrict__ bb,
    float* __restrict__ xout, bf16_t* __restrict__ xbf) {
  int row = blockIdx.x * 8 + (threadIdx.x >> 5);
  int lane = threadIdx.x & 31;
  const float* xr = xin + (size_t)row * 128;
  const float* tr = t + (size_t)row * 128;
  float v[4], s = 0.f, ss = 0.f;
  #pragma unroll
  for (int k = 0; k < 4; k++) { int c = lane + 32 * k; v[k] = xr[c] + tr[c]; s += v[k]; ss += v[k] * v[k]; }
  #pragma unroll
  for (int sh = 16; sh >= 1; sh >>= 1) { s += __shfl_xor(s, sh, 32); ss += __shfl_xor(ss, sh, 32); }
  float mean = s * (1.f / 128.f), var = ss * (1.f / 128.f) - mean * mean;
  float rs = rsqrtf(var + 1e-5f);
  #pragma unroll
  for (int k = 0; k < 4; k++) {
    int c = lane + 32 * k;
    float y = (v[k] - mean) * rs * g[c] + bb[c];
    xout[(size_t)row * 128 + c] = y; xbf[(size_t)row * 128 + c] = f2b(y);
  }
}

// ======================================================================
// Fused edge update: per block = 32 rows (fixed b,i; j-tile).
//   h  = gelu([e_ij,e_ji] @ We0 + be0 + src_i + tgt_j)      (WMMA, K=128)
//   e1 = LN(e + h @ We1 + be1)                              (WMMA, K=256)
//   e2 = LN(e1 + gelu(e1 @ Wem1) @ Wem2 + bem2)             (WMMA)
// All intermediates live in 36KB LDS; B-frags direct from packed weights.
// ======================================================================
__global__ __launch_bounds__(256) void edge_kernel(
    const float* __restrict__ e_in, const bf16_t* __restrict__ ebf_in,
    const float* __restrict__ srcb, const float* __restrict__ tgtb,
    const bf16_t* __restrict__ We0, const float* __restrict__ be0,
    const bf16_t* __restrict__ We1, const float* __restrict__ be1,
    const float* __restrict__ g0, const float* __restrict__ b0,
    const bf16_t* __restrict__ Wem1, const bf16_t* __restrict__ Wem2,
    const float* __restrict__ bem2,
    const float* __restrict__ g1, const float* __restrict__ b1,
    float* __restrict__ e_out, bf16_t* __restrict__ ebf_out) {
  __shared__ __align__(16) bf16_t sA[32 * 128];
  __shared__ __align__(16) bf16_t sH[32 * 256];
  __shared__ __align__(16) float  sE1f[32 * 64];
  __shared__ __align__(16) bf16_t sE1b[32 * 64];

  int blk = blockIdx.x;
  int jt = blk & 7, i = (blk >> 3) & 255, b = blk >> 11;
  int j0 = jt * 32;
  int tid = threadIdx.x, wave = tid >> 5, lane = tid & 31;
  int l15 = lane & 15, kh = (lane >> 4) * 8, rhalf = (lane >> 4) * 8, kb16 = (lane >> 4) * 16;

  if (tid < 32) {   // prefetch residual e rows (used in GEMM2 epilogue)
    const float* er = e_in + ((size_t)((b * Nn + i) * Nn + j0 + tid)) * 64;
    __builtin_prefetch(er, 0, 3);
    __builtin_prefetch(er + 32, 0, 3);
  }

  { // ecat tile: cols 0..63 = e_ij, 64..127 = e_ji
    int idx = tid * 16;
    int r = idx >> 7, c = idx & 127;
    const bf16_t* s;
    if (c < 64) s = ebf_in + ((size_t)((b * Nn + i) * Nn + (j0 + r))) * 64 + c;
    else        s = ebf_in + ((size_t)((b * Nn + (j0 + r)) * Nn + i)) * 64 + (c - 64);
    *(uint4*)&sA[idx]     = *(const uint4*)s;
    *(uint4*)&sA[idx + 8] = *(const uint4*)(s + 8);
  }
  __syncthreads();

  // ---- GEMM1: 32x256 = sA(32x128) @ We0(128x256), epi -> sH (bf16)
  {
    f32x8 acc[2][2];
    #pragma unroll
    for (int a = 0; a < 2; a++)
      #pragma unroll
      for (int c = 0; c < 2; c++) acc[a][c] = zero8();
    #pragma unroll
    for (int k0 = 0; k0 < 128; k0 += 32) {
      FragU af[2], bfr[2];
      #pragma unroll
      for (int fm = 0; fm < 2; fm++) {
        int r = fm * 16 + l15;
        af[fm].q[0] = *(const uint4*)&sA[r * 128 + k0 + kh];
        af[fm].q[1] = *(const uint4*)&sA[r * 128 + k0 + 16 + kh];
        int cc = wave * 32 + fm * 16 + l15;
        const bf16_t* bp = We0 + ((size_t)(k0 >> 5) * 256 + cc) * 32 + kb16;
        bfr[fm].q[0] = *(const uint4*)bp;
        bfr[fm].q[1] = *(const uint4*)(bp + 8);
      }
      #pragma unroll
      for (int fm = 0; fm < 2; fm++)
        #pragma unroll
        for (int fn = 0; fn < 2; fn++)
          acc[fm][fn] = __builtin_amdgcn_wmma_f32_16x16x32_bf16(
              false, af[fm].v, false, bfr[fn].v, (short)0, acc[fm][fn], false, false);
    }
    #pragma unroll
    for (int fm = 0; fm < 2; fm++)
      #pragma unroll
      for (int fn = 0; fn < 2; fn++) {
        int gn = wave * 32 + fn * 16 + l15;
        float bv = be0[gn] + srcb[(size_t)(b * Nn + i) * 256 + gn];
        #pragma unroll
        for (int v = 0; v < 8; v++) {
          int gm = fm * 16 + rhalf + v;
          float val = acc[fm][fn][v] + bv + tgtb[(size_t)(b * Nn + j0 + gm) * 256 + gn];
          sH[gm * 256 + gn] = f2b(gelu_f(val));
        }
      }
  }
  __syncthreads();

  // ---- GEMM2: 32x64 = sH(32x256) @ We1(256x64); residual + bias -> sE1f
  {
    int mi = wave >> 2, nj = wave & 3;
    f32x8 acc = zero8();
    #pragma unroll
    for (int k0 = 0; k0 < 256; k0 += 32) {
      FragU af, bfr;
      int r = mi * 16 + l15;
      af.q[0] = *(const uint4*)&sH[r * 256 + k0 + kh];
      af.q[1] = *(const uint4*)&sH[r * 256 + k0 + 16 + kh];
      int cc = nj * 16 + l15;
      const bf16_t* bp = We1 + ((size_t)(k0 >> 5) * 64 + cc) * 32 + kb16;
      bfr.q[0] = *(const uint4*)bp;
      bfr.q[1] = *(const uint4*)(bp + 8);
      acc = __builtin_amdgcn_wmma_f32_16x16x32_bf16(false, af.v, false, bfr.v, (short)0, acc, false, false);
    }
    int gn = nj * 16 + l15;
    #pragma unroll
    for (int v = 0; v < 8; v++) {
      int gm = mi * 16 + rhalf + v;
      float t1 = e_in[((size_t)(b * Nn + i) * Nn + j0 + gm) * 64 + gn] + acc[v] + be1[gn];
      sE1f[gm * 64 + gn] = t1;
    }
  }
  __syncthreads();

  // ---- LN0 -> sE1f (f32) + sE1b (bf16)
  #pragma unroll
  for (int rr = 0; rr < 4; rr++) {
    int r = wave * 4 + rr;
    float x0 = sE1f[r * 64 + lane], x1 = sE1f[r * 64 + 32 + lane];
    float s = x0 + x1, ss = x0 * x0 + x1 * x1;
    #pragma unroll
    for (int sh = 16; sh >= 1; sh >>= 1) { s += __shfl_xor(s, sh, 32); ss += __shfl_xor(ss, sh, 32); }
    float mean = s * (1.f / 64.f), var = ss * (1.f / 64.f) - mean * mean;
    float rs = rsqrtf(var + 1e-5f);
    float y0 = (x0 - mean) * rs * g0[lane] + b0[lane];
    float y1 = (x1 - mean) * rs * g0[32 + lane] + b0[32 + lane];
    sE1f[r * 64 + lane] = y0;      sE1b[r * 64 + lane] = f2b(y0);
    sE1f[r * 64 + 32 + lane] = y1; sE1b[r * 64 + 32 + lane] = f2b(y1);
  }
  __syncthreads();

  // ---- GEMM3: 32x256 = sE1b(32x64) @ Wem1(64x256), gelu -> sH
  {
    f32x8 acc[2][2];
    #pragma unroll
    for (int a = 0; a < 2; a++)
      #pragma unroll
      for (int c = 0; c < 2; c++) acc[a][c] = zero8();
    #pragma unroll
    for (int k0 = 0; k0 < 64; k0 += 32) {
      FragU af[2], bfr[2];
      #pragma unroll
      for (int fm = 0; fm < 2; fm++) {
        int r = fm * 16 + l15;
        af[fm].q[0] = *(const uint4*)&sE1b[r * 64 + k0 + kh];
        af[fm].q[1] = *(const uint4*)&sE1b[r * 64 + k0 + 16 + kh];
        int cc = wave * 32 + fm * 16 + l15;
        const bf16_t* bp = Wem1 + ((size_t)(k0 >> 5) * 256 + cc) * 32 + kb16;
        bfr[fm].q[0] = *(const uint4*)bp;
        bfr[fm].q[1] = *(const uint4*)(bp + 8);
      }
      #pragma unroll
      for (int fm = 0; fm < 2; fm++)
        #pragma unroll
        for (int fn = 0; fn < 2; fn++)
          acc[fm][fn] = __builtin_amdgcn_wmma_f32_16x16x32_bf16(
              false, af[fm].v, false, bfr[fn].v, (short)0, acc[fm][fn], false, false);
    }
    #pragma unroll
    for (int fm = 0; fm < 2; fm++)
      #pragma unroll
      for (int fn = 0; fn < 2; fn++) {
        int gn = wave * 32 + fn * 16 + l15;
        #pragma unroll
        for (int v = 0; v < 8; v++)
          sH[(fm * 16 + rhalf + v) * 256 + gn] = f2b(gelu_f(acc[fm][fn][v]));
      }
  }
  __syncthreads();

  // ---- GEMM4: 32x64 = sH @ Wem2(256x64); residual e1 + bias -> sA (as f32)
  float* sAf = (float*)sA;
  {
    int mi = wave >> 2, nj = wave & 3;
    f32x8 acc = zero8();
    #pragma unroll
    for (int k0 = 0; k0 < 256; k0 += 32) {
      FragU af, bfr;
      int r = mi * 16 + l15;
      af.q[0] = *(const uint4*)&sH[r * 256 + k0 + kh];
      af.q[1] = *(const uint4*)&sH[r * 256 + k0 + 16 + kh];
      int cc = nj * 16 + l15;
      const bf16_t* bp = Wem2 + ((size_t)(k0 >> 5) * 64 + cc) * 32 + kb16;
      bfr.q[0] = *(const uint4*)bp;
      bfr.q[1] = *(const uint4*)(bp + 8);
      acc = __builtin_amdgcn_wmma_f32_16x16x32_bf16(false, af.v, false, bfr.v, (short)0, acc, false, false);
    }
    int gn = nj * 16 + l15;
    #pragma unroll
    for (int v = 0; v < 8; v++) {
      int gm = mi * 16 + rhalf + v;
      sAf[gm * 64 + gn] = sE1f[gm * 64 + gn] + acc[v] + bem2[gn];
    }
  }
  __syncthreads();

  // ---- LN1 -> global e (f32 + bf16)
  #pragma unroll
  for (int rr = 0; rr < 4; rr++) {
    int r = wave * 4 + rr; int j = j0 + r;
    float x0 = sAf[r * 64 + lane], x1 = sAf[r * 64 + 32 + lane];
    float s = x0 + x1, ss = x0 * x0 + x1 * x1;
    #pragma unroll
    for (int sh = 16; sh >= 1; sh >>= 1) { s += __shfl_xor(s, sh, 32); ss += __shfl_xor(ss, sh, 32); }
    float mean = s * (1.f / 64.f), var = ss * (1.f / 64.f) - mean * mean;
    float rs = rsqrtf(var + 1e-5f);
    float y0 = (x0 - mean) * rs * g1[lane] + b1[lane];
    float y1 = (x1 - mean) * rs * g1[32 + lane] + b1[32 + lane];
    size_t o = ((size_t)(b * Nn + i) * Nn + j) * 64;
    e_out[o + lane] = y0;        e_out[o + 32 + lane] = y1;
    ebf_out[o + lane] = f2b(y0); ebf_out[o + 32 + lane] = f2b(y1);
  }
}

// ======================================================================
// host
// ======================================================================
extern "C" void kernel_launch(void* const* d_in, const int* in_sizes, int n_in,
                              void* d_out, int out_size, void* d_ws, size_t ws_size,
                              hipStream_t stream) {
  (void)in_sizes; (void)n_in; (void)out_size; (void)ws_size;

  // -------- workspace carve --------
  char* w = (char*)d_ws; size_t off = 0;
  auto alloc = [&](size_t bytes) -> char* {
    char* p = w + off; off = (off + bytes + 255) & ~(size_t)255; return p;
  };
  const size_t NE = (size_t)Bn * Nn * Nn * DEc;   // 8388608
  float*  e_ping  = (float*)alloc(NE * 4);
  float*  e_pong  = (float*)alloc(NE * 4);
  bf16_t* ebf0    = (bf16_t*)alloc(NE * 2);
  bf16_t* ebf1    = (bf16_t*)alloc(NE * 2);
  bf16_t* wbf     = (bf16_t*)alloc((size_t)786432 * 2);
  float*  xbuf    = (float*)alloc(65536 * 4);
  bf16_t* xbf     = (bf16_t*)alloc(65536 * 2);
  float*  qkvn    = (float*)alloc(196608 * 4);
  float*  aout    = (float*)alloc(65536 * 4);
  bf16_t* aoutbf  = (bf16_t*)alloc(65536 * 2);
  float*  attn1   = (float*)alloc(65536 * 4);
  bf16_t* attn1bf = (bf16_t*)alloc(65536 * 2);
  float*  tbuf    = (float*)alloc(65536 * 4);
  float*  h1      = (float*)alloc(262144 * 4);
  bf16_t* h1bf    = (bf16_t*)alloc(262144 * 2);
  float*  srcb    = (float*)alloc(131072 * 4);
  float*  tgtb    = (float*)alloc(131072 * 4);

  // bf16 (frag-packed) weight sub-offsets (elements)
  const size_t OFF_QKVN = 0,      OFF_QKVE = 98304, OFF_WO = 163840, OFF_WL0 = 196608,
               OFF_WM1 = 229376,  OFF_WM2 = 360448, OFF_WE0 = 491520, OFF_WS = 557056,
               OFF_WT = 622592,   OFF_WE1 = 688128, OFF_WEM1 = 720896, OFF_WEM2 = 753664;

  auto pack = [&](const void* s, bf16_t* d, int K, int N) {
    dim3 grid((K * N + 255) / 256, Lc);
    repack_b<<<grid, 256, 0, stream>>>((const float*)s, d, K, N);
  };
  pack(d_in[2],  wbf + OFF_QKVN, DNc, 3 * DHc);
  pack(d_in[3],  wbf + OFF_QKVE, DEc, 4 * DHc);
  pack(d_in[4],  wbf + OFF_WO,   DHc, DNc);
  pack(d_in[6],  wbf + OFF_WL0,  DNc, DNc);
  pack(d_in[10], wbf + OFF_WM1,  DNc, DNHc);
  pack(d_in[11], wbf + OFF_WM2,  DNHc, DNc);
  pack(d_in[15], wbf + OFF_WE0,  2 * DEc, DEHc);
  pack(d_in[17], wbf + OFF_WS,   DNc, DEHc);
  pack(d_in[19], wbf + OFF_WT,   DNc, DEHc);
  pack(d_in[21], wbf + OFF_WE1,  DEHc, DEc);
  pack(d_in[25], wbf + OFF_WEM1, DEc, DEHc);
  pack(d_in[26], wbf + OFF_WEM2, DEHc, DEc);
  cvt_f32_bf16<<<(Bn * Nn * DNc + 255) / 256, 256, 0, stream>>>((const float*)d_in[0], xbf, Bn * Nn * DNc);
  cvt_f32_bf16<<<((int)NE + 255) / 256, 256, 0, stream>>>((const float*)d_in[1], ebf0, (int)NE);

  const float* cur_x = (const float*)d_in[0];
  const float* cur_e = (const float*)d_in[1];
  bf16_t* cur_ebf = ebf0;

  for (int l = 0; l < Lc; l++) {
    const bf16_t* wqkvn_l = wbf + OFF_QKVN + (size_t)l * DNc * 3 * DHc;
    const bf16_t* wqkve_l = wbf + OFF_QKVE + (size_t)l * DEc * 4 * DHc;
    const bf16_t* wo_l    = wbf + OFF_WO   + (size_t)l * DHc * DNc;
    const bf16_t* wl0_l   = wbf + OFF_WL0  + (size_t)l * DNc * DNc;
    const bf16_t* wm1_l   = wbf + OFF_WM1  + (size_t)l * DNc * DNHc;
    const bf16_t* wm2_l   = wbf + OFF_WM2  + (size_t)l * DNHc * DNc;
    const bf16_t* we0_l   = wbf + OFF_WE0  + (size_t)l * 2 * DEc * DEHc;
    const bf16_t* ws_l    = wbf + OFF_WS   + (size_t)l * DNc * DEHc;
    const bf16_t* wt_l    = wbf + OFF_WT   + (size_t)l * DNc * DEHc;
    const bf16_t* we1_l   = wbf + OFF_WE1  + (size_t)l * DEHc * DEc;
    const bf16_t* wem1_l  = wbf + OFF_WEM1 + (size_t)l * DEc * DEHc;
    const bf16_t* wem2_l  = wbf + OFF_WEM2 + (size_t)l * DEHc * DEc;
    const float* bo_l   = (const float*)d_in[5]  + (size_t)l * DNc;
    const float* bl0_l  = (const float*)d_in[7]  + (size_t)l * DNc;
    const float* ln0g_l = (const float*)d_in[8]  + (size_t)l * DNc;
    const float* ln0b_l = (const float*)d_in[9]  + (size_t)l * DNc;
    const float* bm2_l  = (const float*)d_in[12] + (size_t)l * DNc;
    const float* ln1g_l = (const float*)d_in[13] + (size_t)l * DNc;
    const float* ln1b_l = (const float*)d_in[14] + (size_t)l * DNc;
    const float* be0_l  = (const float*)d_in[16] + (size_t)l * DEHc;
    const float* bs_l   = (const float*)d_in[18] + (size_t)l * DEHc;
    const float* bt_l   = (const float*)d_in[20] + (size_t)l * DEHc;
    const float* be1_l  = (const float*)d_in[22] + (size_t)l * DEc;
    const float* e0g_l  = (const float*)d_in[23] + (size_t)l * DEc;
    const float* e0b_l  = (const float*)d_in[24] + (size_t)l * DEc;
    const float* bem2_l = (const float*)d_in[27] + (size_t)l * DEc;
    const float* e1g_l  = (const float*)d_in[28] + (size_t)l * DEc;
    const float* e1b_l  = (const float*)d_in[29] + (size_t)l * DEc;

    // 1. node QKV: (512x384) = xbf @ Wqkv_n
    gemm_bf16<0, false><<<dim3(8, 6), 128, 0, stream>>>(xbf, wqkvn_l, nullptr, qkvn, nullptr, 512, 384, 128);
    // 2. fused edge-QKV projection + flash attention
    attn_kernel<<<Bn * Nn, 256, 0, stream>>>(cur_ebf, wqkve_l, qkvn, aout, aoutbf);
    // 3. out @ Wo + bo
    gemm_bf16<1, true><<<dim3(8, 2), 128, 0, stream>>>(aoutbf, wo_l, bo_l, attn1, attn1bf, 512, 128, 128);
    // 4. attn_out @ Wl0 + bl0
    gemm_bf16<1, false><<<dim3(8, 2), 128, 0, stream>>>(attn1bf, wl0_l, bl0_l, tbuf, nullptr, 512, 128, 128);
    // 5. x = LN(x + t)
    addln_node<<<64, 256, 0, stream>>>(cur_x, tbuf, ln0g_l, ln0b_l, xbuf, xbf);
    cur_x = xbuf;
    // 6. h1 = gelu(x @ Wm1)
    gemm_bf16<2, true><<<dim3(8, 8), 128, 0, stream>>>(xbf, wm1_l, nullptr, h1, h1bf, 512, 512, 128);
    // 7. t = h1 @ Wm2 + bm2
    gemm_bf16<1, false><<<dim3(8, 2), 128, 0, stream>>>(h1bf, wm2_l, bm2_l, tbuf, nullptr, 512, 128, 512);
    // 8. x = LN(x + t)
    addln_node<<<64, 256, 0, stream>>>(cur_x, tbuf, ln1g_l, ln1b_l, xbuf, xbf);
    // 9/10. src, tgt
    gemm_bf16<1, false><<<dim3(8, 4), 128, 0, stream>>>(xbf, ws_l, bs_l, srcb, nullptr, 512, 256, 128);
    gemm_bf16<1, false><<<dim3(8, 4), 128, 0, stream>>>(xbf, wt_l, bt_l, tgtb, nullptr, 512, 256, 128);
    // 11. fused edge update
    float*  e_next   = (l & 1) ? e_pong : e_ping;
    bf16_t* ebf_next = (l & 1) ? ebf0 : ebf1;
    edge_kernel<<<Bn * Nn * (Nn / 32), 256, 0, stream>>>(
        cur_e, cur_ebf, srcb, tgtb, we0_l, be0_l, we1_l, be1_l, e0g_l, e0b_l,
        wem1_l, wem2_l, bem2_l, e1g_l, e1b_l, e_next, ebf_next);
    cur_e = e_next; cur_ebf = ebf_next;
  }

  float* outp = (float*)d_out;
  (void)hipMemcpyAsync(outp, cur_x, (size_t)Bn * Nn * DNc * 4, hipMemcpyDeviceToDevice, stream);
  (void)hipMemcpyAsync(outp + (size_t)Bn * Nn * DNc, cur_e, NE * 4, hipMemcpyDeviceToDevice, stream);
}
